// EoMTLayer_28501402977032
// MI455X (gfx1250) — matrix operand order repaired and checked by reading, not compiled, not run on hardware
//
#include <hip/hip_runtime.h>
#include <hip/hip_bf16.h>

// ---------------- problem constants ----------------
#define B_   8
#define S_   1124          // SEQ_LEN
#define SP   1152          // padded seq (multiple of 32; 1152 = 9*128)
#define E_   1024
#define H_   16
#define D_   64
#define NQ   100
#define NP   1024
#define MLP_ 4096
#define NEGINF (-1.0e9f)

typedef __bf16 v16bf __attribute__((ext_vector_type(16)));
typedef __bf16 v8bf  __attribute__((ext_vector_type(8)));
typedef float  v8f   __attribute__((ext_vector_type(8)));

union Frag { v16bf v16; v8bf v8[2]; };

static __device__ inline v8f wmma_bf16(v16bf a, v16bf b, v8f c) {
  // D(16x16,f32) = A(16x32,bf16) * B(32x16,bf16) + C
  return __builtin_amdgcn_wmma_f32_16x16x32_bf16(false, a, false, b, (short)0, c, false, false);
}
static __device__ inline v8f vzero8() {
  v8f z;
#pragma unroll
  for (int i = 0; i < 8; i++) z[i] = 0.f;
  return z;
}

// ---- CDNA5 async global->LDS copy (ASYNCcnt-tracked, bypasses VGPRs) ----
// saddr form: global_load_async_to_lds_b128 vdst(LDS byte addr), vaddr(u32 off), s[base]
static __device__ inline void async_load_b128(unsigned lds_addr, unsigned voff,
                                              unsigned long long sbase) {
  asm volatile("global_load_async_to_lds_b128 %0, %1, %2 offset:0"
               :: "v"(lds_addr), "v"(voff), "s"(sbase)
               : "memory");
}
template <int N> static __device__ inline void wait_asynccnt() {
  asm volatile("s_wait_asynccnt %0" :: "n"(N) : "memory");
}

// ---------------- utility kernels ----------------
__global__ void zero_kernel(unsigned int* p, size_t nwords) {
  for (size_t i = (size_t)blockIdx.x * blockDim.x + threadIdx.x; i < nwords;
       i += (size_t)gridDim.x * blockDim.x)
    p[i] = 0u;
}

// W[K][N] f32  ->  Wt[N][K] bf16 (LDS tiled transpose, 32x32 tiles)
__global__ __launch_bounds__(256) void wconv(const float* __restrict__ W,
                                             __bf16* __restrict__ Wt, int K, int N) {
  __shared__ float t[32][33];
  const int n0 = blockIdx.x * 32, k0 = blockIdx.y * 32;
  const int tx = threadIdx.x, ty = threadIdx.y;   // (32,8)
#pragma unroll
  for (int i = 0; i < 4; i++)
    t[ty + i * 8][tx] = W[(size_t)(k0 + ty + i * 8) * N + n0 + tx];
  __syncthreads();
#pragma unroll
  for (int i = 0; i < 4; i++)
    Wt[(size_t)(n0 + ty + i * 8) * K + k0 + tx] = (__bf16)t[tx][ty + i * 8];
}

// LayerNorm over last dim (1024); writes padded bf16 buffer (zeros in pad rows).
__global__ __launch_bounds__(256) void ln_kernel(const float* __restrict__ in, int inPadded,
                                                 const float* __restrict__ g,
                                                 const float* __restrict__ beta,
                                                 __bf16* __restrict__ out) {
  const int row = blockIdx.x;
  const int b = row / SP, s = row % SP;
  const int tid = threadIdx.x;
  __bf16* orow = out + (size_t)row * E_;
  if (s >= S_) {
    for (int e = tid; e < E_; e += 256) orow[e] = (__bf16)0.0f;
    return;
  }
  const float* irow = inPadded ? (in + (size_t)row * E_) : (in + ((size_t)b * S_ + s) * E_);
  float vals[4], sum = 0.f, sq = 0.f;
#pragma unroll
  for (int kx = 0; kx < 4; kx++) {
    float v = irow[tid + kx * 256];
    vals[kx] = v; sum += v; sq += v * v;
  }
  __shared__ float rs[256], rq[256];
  rs[tid] = sum; rq[tid] = sq;
  __syncthreads();
  for (int off = 128; off > 0; off >>= 1) {
    if (tid < off) { rs[tid] += rs[tid + off]; rq[tid] += rq[tid + off]; }
    __syncthreads();
  }
  const float mean = rs[0] * (1.f / E_);
  const float var  = rq[0] * (1.f / E_) - mean * mean;
  const float rstd = rsqrtf(var + 1e-6f);
#pragma unroll
  for (int kx = 0; kx < 4; kx++) {
    int e = tid + kx * 256;
    orow[e] = (__bf16)((vals[kx] - mean) * rstd * g[e] + beta[e]);
  }
}

// ---------------- WMMA GEMM with async double-buffered LDS staging ----------------
// C[M,N] = A[M,K](bf16,row) * Wt[N,K](bf16,row)^T
// Block 256 thr = 8 waves (2x4); 64x64 tile/wave; block tile 128x256; K step 32.
// A tile 128x32 / B tile 256x32 staged via GLOBAL_LOAD_ASYNC_TO_LDS_B128, 2 buffers.
// LDS row stride 40 elems (80B) keeps ds_load_b128 fragment reads bank-spread.
#define LDS_STRIDE 40
#define A_ELEMS    (128 * LDS_STRIDE)           // 5120 elems
#define AB_ELEMS   (A_ELEMS + 256 * LDS_STRIDE) // 15360 elems per buffer
// EPI 0: QKV scatter -> q,k,vT (q pre-scaled by 1/sqrt(D))
// EPI 1: + bias + x(unpadded) residual -> x2 (f32, padded)
// EPI 2: + bias, exact GELU -> bf16
// EPI 3: + bias + x2(padded) residual -> d_out (f32, unpadded)
template <int EPI>
__global__ __launch_bounds__(256) void gemm_bf16_kernel(
    const __bf16* __restrict__ A, const __bf16* __restrict__ Wt, int K,
    const float* __restrict__ bias, const float* __restrict__ res,
    float* __restrict__ fout, __bf16* __restrict__ bout,
    __bf16* __restrict__ qo, __bf16* __restrict__ ko, __bf16* __restrict__ vto) {
  const int tid  = threadIdx.x;
  const int lane = tid & 31, wid = tid >> 5;
  const int hlf  = lane >> 4, ln = lane & 15;
  const int wm   = wid >> 2,  wn = wid & 3;
  const int rowBlock = blockIdx.y;               // 0..71 ; 9 blocks per batch
  const int b  = rowBlock / 9;
  const int s0 = (rowBlock % 9) * 128 + wm * 64;
  const int CN = blockIdx.x * 256 + wn * 64;

  __shared__ __align__(16) __bf16 smem[2][AB_ELEMS];

  // per-thread chunk geometry (16B chunks): A = 512 chunks, B = 1024 chunks
  auto issue_tile = [&](int kt, int buf) {
    const unsigned ldsA = (unsigned)(uintptr_t)&smem[buf][0];
    const unsigned ldsB = (unsigned)(uintptr_t)&smem[buf][A_ELEMS];
    const unsigned long long baseA =
        (unsigned long long)(uintptr_t)(A + (size_t)rowBlock * 128 * K + kt);
    const unsigned long long baseB =
        (unsigned long long)(uintptr_t)(Wt + (size_t)blockIdx.x * 256 * K + kt);
#pragma unroll
    for (int j = 0; j < 2; ++j) {               // 2 async ops/wave for A
      const int c = tid + 256 * j, row = c >> 2, o = c & 3;
      async_load_b128(ldsA + row * (LDS_STRIDE * 2) + o * 16,
                      (unsigned)((unsigned)row * (unsigned)K * 2u + o * 16u), baseA);
    }
#pragma unroll
    for (int j = 0; j < 4; ++j) {               // 4 async ops/wave for B
      const int c = tid + 256 * j, row = c >> 2, o = c & 3;
      async_load_b128(ldsB + row * (LDS_STRIDE * 2) + o * 16,
                      (unsigned)((unsigned)row * (unsigned)K * 2u + o * 16u), baseB);
    }
  };

  v8f acc[4][4];
#pragma unroll
  for (int i = 0; i < 4; i++)
#pragma unroll
    for (int j = 0; j < 4; j++) acc[i][j] = vzero8();

  const int T = K / 32;
  issue_tile(0, 0);
  for (int it = 0; it < T; ++it) {
    const int cur = it & 1;
    if (it + 1 < T) {
      issue_tile((it + 1) * 32, cur ^ 1);       // prefetch next tile (other buffer)
      wait_asynccnt<6>();                       // oldest 6 (current tile) done
    } else {
      wait_asynccnt<0>();
    }
    __syncthreads();                            // current tile visible to all waves

    const __bf16* As = &smem[cur][0];
    const __bf16* Bs = &smem[cur][A_ELEMS];
    Frag am[4];
#pragma unroll
    for (int i = 0; i < 4; i++) {
      // A 16x32 bf16 frag: lanes 0-15 K {0..15}, lanes 16-31 K {16..31}
      const __bf16* p = As + (wm * 64 + i * 16 + ln) * LDS_STRIDE + hlf * 16;
      am[i].v8[0] = *(const v8bf*)p;
      am[i].v8[1] = *(const v8bf*)(p + 8);
    }
    Frag bn[4];
#pragma unroll
    for (int j = 0; j < 4; j++) {
      // B 32x16 bf16 frag: lane = col n, 16 contiguous K at hlf*16
      const __bf16* p = Bs + (wn * 64 + j * 16 + ln) * LDS_STRIDE + hlf * 16;
      bn[j].v8[0] = *(const v8bf*)p;
      bn[j].v8[1] = *(const v8bf*)(p + 8);
    }
#pragma unroll
    for (int i = 0; i < 4; i++)
#pragma unroll
      for (int j = 0; j < 4; j++)
        acc[i][j] = wmma_bf16(am[i].v16, bn[j].v16, acc[i][j]);
    __syncthreads();                            // reads done before buffer is re-filled
  }

#pragma unroll
  for (int i = 0; i < 4; i++)
#pragma unroll
    for (int j = 0; j < 4; j++)
#pragma unroll
      for (int r = 0; r < 8; r++) {
        const int s = s0 + i * 16 + hlf * 8 + r;   // D layout: M = r + 8*(lane/16)
        if (s >= S_) continue;
        const int col = CN + j * 16 + ln;          // N = lane%16
        const float v = acc[i][j][r];
        if constexpr (EPI == 0) {
          const int which = col >> 10, hc = col & 1023, hh = hc >> 6, d = hc & 63;
          if (which == 0)
            qo[(((size_t)b * H_ + hh) * SP + s) * D_ + d] = (__bf16)(v * 0.125f); // *SCALE
          else if (which == 1)
            ko[(((size_t)b * H_ + hh) * SP + s) * D_ + d] = (__bf16)v;
          else
            vto[(((size_t)b * H_ + hh) * D_ + d) * SP + s] = (__bf16)v;           // V^T
        } else if constexpr (EPI == 1) {
          fout[((size_t)b * SP + s) * E_ + col] =
              v + bias[col] + res[((size_t)b * S_ + s) * E_ + col];
        } else if constexpr (EPI == 2) {
          const float u = v + bias[col];
          bout[((size_t)b * SP + s) * (size_t)MLP_ + col] =
              (__bf16)(0.5f * u * (1.f + erff(u * 0.70710678118654752f)));
        } else {
          fout[((size_t)b * S_ + s) * E_ + col] =
              v + bias[col] + res[((size_t)b * SP + s) * E_ + col];
        }
      }
}

// ---------------- fused flash attention, one wave per 16-query tile ----------------
__global__ __launch_bounds__(32) void attn_kernel(const __bf16* __restrict__ q,
                                                  const __bf16* __restrict__ k,
                                                  const __bf16* __restrict__ vT,
                                                  const float* __restrict__ mask,
                                                  __bf16* __restrict__ ctx) {
  const int lane = threadIdx.x;
  const int hlf = lane >> 4, ln = lane & 15;
  const int qt = blockIdx.x, h = blockIdx.y, b = blockIdx.z;
  const size_t bh = (size_t)b * H_ + h;
  const __bf16* qb = q  + bh * SP * D_;
  const __bf16* kb = k  + bh * SP * D_;
  const __bf16* vb = vT + bh * D_ * SP;

  __shared__ __align__(16) __bf16 pls[16][40];   // 16 queries x 32 keys, padded rows

  // Q A-fragments for head-dim chunks [0,32) and [32,64); rows = qt*16 + ln
  Frag aq0, aq1;
  {
    const __bf16* p = qb + (size_t)(qt * 16 + ln) * D_ + hlf * 16;
    aq0.v8[0] = *(const v8bf*)p;        aq0.v8[1] = *(const v8bf*)(p + 8);
    aq1.v8[0] = *(const v8bf*)(p + 32); aq1.v8[1] = *(const v8bf*)(p + 40);
  }

  float mrow[8], lrow[8];
  v8f o[4];
#pragma unroll
  for (int r = 0; r < 8; r++) { mrow[r] = -1e30f; lrow[r] = 0.f; }
#pragma unroll
  for (int j = 0; j < 4; j++) o[j] = vzero8();

  const bool qmask = (qt >= 64);   // tiles 64..70 hold query rows >= 1024

  for (int kt = 0; kt < SP; kt += 32) {
    // ---- scores: two 16x16 tiles over 32 keys, contraction over head dim 64 ----
    v8f sc[2];
#pragma unroll
    for (int t = 0; t < 2; t++) {
      sc[t] = vzero8();
      const __bf16* kp = kb + (size_t)(kt + t * 16 + ln) * D_ + hlf * 16;
      v16bf bk0 = *(const v16bf*)kp;          // K^T fragment, dims 0..31
      v16bf bk1 = *(const v16bf*)(kp + 32);   // dims 32..63
      sc[t] = wmma_bf16(aq0.v16, bk0, sc[t]);
      sc[t] = wmma_bf16(aq1.v16, bk1, sc[t]);
    }
    // ---- mask ----
#pragma unroll
    for (int t = 0; t < 2; t++) {
      const int key = kt + t * 16 + ln;
      const bool kvalid = key < S_;
#pragma unroll
      for (int r = 0; r < 8; r++) {
        float v = sc[t][r];
        if (!kvalid) v = NEGINF;
        else if (qmask && key < NP) {
          const int qi = qt * 16 + hlf * 8 + r;   // >= 1024 here
          if (qi < S_) {
            const float mv = mask[((size_t)b * NQ + (qi - NP)) * (size_t)NP + key];
            if (mv <= 0.5f) v = NEGINF;
          }
        }
        sc[t][r] = v;
      }
    }
    // ---- online softmax (row stats reduced across 16-lane half-groups) ----
    float mx[8];
#pragma unroll
    for (int r = 0; r < 8; r++) mx[r] = fmaxf(sc[0][r], sc[1][r]);
    for (int off = 1; off < 16; off <<= 1)
#pragma unroll
      for (int r = 0; r < 8; r++) mx[r] = fmaxf(mx[r], __shfl_xor(mx[r], off, 32));
    float corr[8];
#pragma unroll
    for (int r = 0; r < 8; r++) {
      const float mn = fmaxf(mrow[r], mx[r]);
      corr[r] = __expf(mrow[r] - mn);
      mrow[r] = mn;
    }
#pragma unroll
    for (int t = 0; t < 2; t++)
#pragma unroll
      for (int r = 0; r < 8; r++) sc[t][r] = __expf(sc[t][r] - mrow[r]);
    float rsum[8];
#pragma unroll
    for (int r = 0; r < 8; r++) rsum[r] = sc[0][r] + sc[1][r];
    for (int off = 1; off < 16; off <<= 1)
#pragma unroll
      for (int r = 0; r < 8; r++) rsum[r] += __shfl_xor(rsum[r], off, 32);
#pragma unroll
    for (int r = 0; r < 8; r++) lrow[r] = lrow[r] * corr[r] + rsum[r];
#pragma unroll
    for (int j = 0; j < 4; j++)
#pragma unroll
      for (int r = 0; r < 8; r++) o[j][r] *= corr[r];

    // ---- transpose P (D layout -> A layout) through LDS ----
#pragma unroll
    for (int t = 0; t < 2; t++)
#pragma unroll
      for (int r = 0; r < 8; r++)
        pls[hlf * 8 + r][t * 16 + ln] = (__bf16)sc[t][r];
    __syncthreads();
    Frag ap;
    {
      const __bf16* pp = &pls[ln][hlf * 16];
      ap.v8[0] = *(const v8bf*)pp;
      ap.v8[1] = *(const v8bf*)(pp + 8);
    }
    // ---- O += P(16x32) * V(32x64), via V^T rows (key-contiguous) ----
#pragma unroll
    for (int j = 0; j < 4; j++) {
      v16bf bv = *(const v16bf*)(vb + (size_t)(j * 16 + ln) * SP + kt + hlf * 16);
      o[j] = wmma_bf16(ap.v16, bv, o[j]);
    }
    __syncthreads();
  }

  // ---- finalize + write ctx[b][s][h*64+d] ----
#pragma unroll
  for (int j = 0; j < 4; j++)
#pragma unroll
    for (int r = 0; r < 8; r++) {
      const int s = qt * 16 + hlf * 8 + r;
      if (s < S_) {
        const float val = o[j][r] / lrow[r];
        ctx[((size_t)b * SP + s) * E_ + h * D_ + j * 16 + ln] = (__bf16)val;
      }
    }
}

// ---------------- host-side orchestration ----------------
extern "C" void kernel_launch(void* const* d_in, const int* in_sizes, int n_in,
                              void* d_out, int out_size, void* d_ws, size_t ws_size,
                              hipStream_t stream) {
  (void)in_sizes; (void)n_in; (void)out_size; (void)ws_size;
  const float* x      = (const float*)d_in[0];
  const float* mask   = (const float*)d_in[1];
  const float* qkv_w  = (const float*)d_in[2];
  const float* proj_w = (const float*)d_in[3];
  const float* proj_b = (const float*)d_in[4];
  const float* ln1_g  = (const float*)d_in[5];
  const float* ln1_b  = (const float*)d_in[6];
  const float* ln2_g  = (const float*)d_in[7];
  const float* ln2_b  = (const float*)d_in[8];
  const float* fc1_w  = (const float*)d_in[9];
  const float* fc1_b  = (const float*)d_in[10];
  const float* fc2_w  = (const float*)d_in[11];
  const float* fc2_b  = (const float*)d_in[12];
  float* out = (float*)d_out;

  char* w = (char*)d_ws;
  auto take = [&](size_t bytes) { char* p = w; w += (bytes + 255) & ~(size_t)255; return p; };
  __bf16* xn    = (__bf16*)take((size_t)B_ * SP * E_ * 2);     // LN1 out; reused as LN2 out
  __bf16* qkvT  = (__bf16*)take((size_t)3 * E_ * E_ * 2);
  __bf16* projT = (__bf16*)take((size_t)E_ * E_ * 2);
  __bf16* fc1T  = (__bf16*)take((size_t)E_ * MLP_ * 2);
  __bf16* fc2T  = (__bf16*)take((size_t)MLP_ * E_ * 2);
  __bf16* qb    = (__bf16*)take((size_t)B_ * H_ * SP * D_ * 2);
  __bf16* kb    = (__bf16*)take((size_t)B_ * H_ * SP * D_ * 2);
  __bf16* vT    = (__bf16*)take((size_t)B_ * H_ * D_ * SP * 2);
  __bf16* ctx   = (__bf16*)take((size_t)B_ * SP * E_ * 2);
  float*  x2    = (float*) take((size_t)B_ * SP * E_ * 4);
  __bf16* gbuf  = (__bf16*)take((size_t)B_ * SP * MLP_ * 2);

  // 1) weights -> bf16, transposed to [N][K] (K-contiguous WMMA-B fragments)
  wconv<<<dim3(3 * E_ / 32, E_ / 32), dim3(32, 8), 0, stream>>>(qkv_w, qkvT, E_, 3 * E_);
  wconv<<<dim3(E_ / 32, E_ / 32),     dim3(32, 8), 0, stream>>>(proj_w, projT, E_, E_);
  wconv<<<dim3(MLP_ / 32, E_ / 32),   dim3(32, 8), 0, stream>>>(fc1_w, fc1T, E_, MLP_);
  wconv<<<dim3(E_ / 32, MLP_ / 32),   dim3(32, 8), 0, stream>>>(fc2_w, fc2T, MLP_, E_);
  // 2) zero V^T so padded key columns contribute exact zeros to P*V
  zero_kernel<<<4096, 256, 0, stream>>>((unsigned int*)vT, (size_t)B_ * H_ * D_ * SP / 2);
  // 3) LN1 -> xn (bf16, padded rows zeroed)
  ln_kernel<<<B_ * SP, 256, 0, stream>>>(x, 0, ln1_g, ln1_b, xn);
  // 4) QKV GEMM, scatter into q (pre-scaled), k, V^T
  gemm_bf16_kernel<0><<<dim3(3 * E_ / 256, 72), 256, 0, stream>>>(
      xn, qkvT, E_, nullptr, nullptr, nullptr, nullptr, qb, kb, vT);
  // 5) fused masked flash attention -> ctx
  attn_kernel<<<dim3(71, H_, B_), 32, 0, stream>>>(qb, kb, vT, mask, ctx);
  // 6) proj + bias + residual(x) -> x2
  gemm_bf16_kernel<1><<<dim3(E_ / 256, 72), 256, 0, stream>>>(
      ctx, projT, E_, proj_b, x, x2, nullptr, nullptr, nullptr, nullptr);
  // 7) LN2 -> h (reuse xn)
  ln_kernel<<<B_ * SP, 256, 0, stream>>>(x2, 1, ln2_g, ln2_b, xn);
  // 8) FC1 + bias + exact GELU -> gbuf
  gemm_bf16_kernel<2><<<dim3(MLP_ / 256, 72), 256, 0, stream>>>(
      xn, fc1T, E_, fc1_b, nullptr, nullptr, gbuf, nullptr, nullptr, nullptr);
  // 9) FC2 + bias + residual(x2) -> d_out
  gemm_bf16_kernel<3><<<dim3(E_ / 256, 72), 256, 0, stream>>>(
      gbuf, fc2T, MLP_, fc2_b, x2, out, nullptr, nullptr, nullptr, nullptr);
}
